// SelfAttention_8504035246726
// MI455X (gfx1250) — matrix operand order repaired and checked by reading, not compiled, not run on hardware
//
#include <hip/hip_runtime.h>
#include <math.h>

typedef __attribute__((ext_vector_type(16))) _Float16 v16h;
typedef __attribute__((ext_vector_type(8)))  _Float16 v8h;
typedef __attribute__((ext_vector_type(8)))  float    v8f;

#define WMMA(a, b, c) __builtin_amdgcn_wmma_f32_16x16x32_f16( \
    false, (a), false, (b), (short)0, (c), false, false)

constexpr int   Bsz   = 4;
constexpr int   Sq    = 2048;
constexpr int   Dm    = 1024;
constexpr int   Hn    = 16;
constexpr int   DH    = 64;
constexpr float SCALE = 0.125f;   // 1/sqrt(64)
constexpr int   Mtot  = Bsz * Sq; // 8192

// ---------------------------------------------------------------------------
// CDNA5 async global->LDS copy (16 bytes), tracked by ASYNCcnt.
// vdst = LDS byte address (low 32 bits of generic LDS pointer), vaddr = 64-bit
// global address, GV mode. Completion enforced with s_wait_asynccnt.
// ---------------------------------------------------------------------------
__device__ inline void async_copy_b128(const _Float16* g, _Float16* l) {
  asm volatile("global_load_async_to_lds_b128 %0, %1, off"
               :: "v"((unsigned)(size_t)l), "v"(g)
               : "memory");
}
#define WAIT_ASYNC(imm) asm volatile("s_wait_asynccnt " #imm ::: "memory")

// ---------------------------------------------------------------------------
// Kernel 1: LayerNorm (fp32) -> f16, one 256-thread block per token
// ---------------------------------------------------------------------------
__global__ void ln_cast_kernel(const float* __restrict__ x,
                               const float* __restrict__ gamma,
                               const float* __restrict__ beta,
                               _Float16* __restrict__ out) {
  const int t   = blockIdx.x;   // token 0..8191
  const int tid = threadIdx.x;  // 0..255 (4 elems each)
  const float4 v = ((const float4*)(x + (size_t)t * Dm))[tid];
  float s  = v.x + v.y + v.z + v.w;
  float ss = v.x * v.x + v.y * v.y + v.z * v.z + v.w * v.w;
  __shared__ float r1[256];
  __shared__ float r2[256];
  r1[tid] = s; r2[tid] = ss;
  __syncthreads();
  for (int o = 128; o > 0; o >>= 1) {
    if (tid < o) { r1[tid] += r1[tid + o]; r2[tid] += r2[tid + o]; }
    __syncthreads();
  }
  const float mean = r1[0] * (1.0f / Dm);
  const float var  = r2[0] * (1.0f / Dm) - mean * mean;
  const float rstd = rsqrtf(var + 1e-5f);
  const float4 g = ((const float4*)gamma)[tid];
  const float4 b = ((const float4*)beta)[tid];
  _Float16* o4 = out + (size_t)t * Dm + tid * 4;
  o4[0] = (_Float16)((v.x - mean) * rstd * g.x + b.x);
  o4[1] = (_Float16)((v.y - mean) * rstd * g.y + b.y);
  o4[2] = (_Float16)((v.z - mean) * rstd * g.z + b.z);
  o4[3] = (_Float16)((v.w - mean) * rstd * g.w + b.w);
}

// ---------------------------------------------------------------------------
// Kernel 2: fp32 -> f16 weight cast
// ---------------------------------------------------------------------------
__global__ void cast_f32_f16(const float* __restrict__ src,
                             _Float16* __restrict__ dst, int n) {
  int i = blockIdx.x * blockDim.x + threadIdx.x;
  if (i < n) dst[i] = (_Float16)src[i];
}

// ---------------------------------------------------------------------------
// Shared 128x128 WMMA GEMM tile, K-step 64, double-buffered LDS with
// async global->LDS staging. Uniform software pipeline: every iteration
// stages the next slab (index wraps on the last iteration so the loop body
// is branch-free) and waits for the oldest 8 async ops before consuming.
// A row-major [M x 1024], B row-major [1024 x N].
// 256 threads = 8 waves; wave (wm 0..3, wn 0..1) computes 32x64.
// ---------------------------------------------------------------------------
__device__ inline void gemm_tile_128x128(const _Float16* __restrict__ A,
                                         const _Float16* __restrict__ Bm,
                                         int blockM, int blockN,
                                         v8f (&acc)[2][4]) {
  __shared__ __attribute__((aligned(32))) _Float16 sA[2][128 * 64]; // 2 x 16 KB
  __shared__ __attribute__((aligned(32))) _Float16 sB[2][64 * 128]; // 2 x 16 KB

  const int tid  = threadIdx.x;
  const int lane = tid & 31;
  const int wid  = tid >> 5;
  const int wm   = wid & 3;        // 4 waves along M (32 rows each)
  const int wn   = wid >> 2;       // 2 waves along N (64 cols each)
  const int half = lane >> 4;
  const int l16  = lane & 15;
  const int k0l  = half * 8;       // A-frag K base per ISA layout

  // staging assignment: 8 x b128 async copies per thread per stage
  const int arow = tid & 127, acol = (tid >> 7) * 32;  // A: 128 x 64 halfs
  const int brow = tid >> 2,  bcol = (tid & 3) * 32;   // B:  64 x 128 halfs

  auto stage = [&](int buf, int k0) {
    const _Float16* ag = &A[(size_t)(blockM + arow) * Dm + k0 + acol];
    _Float16*       al = &sA[buf][arow * 64 + acol];
    const _Float16* bg = &Bm[(size_t)(k0 + brow) * Dm + blockN + bcol];
    _Float16*       bl = &sB[buf][brow * 128 + bcol];
#pragma unroll
    for (int u = 0; u < 4; ++u) {
      async_copy_b128(ag + u * 8, al + u * 8);
      async_copy_b128(bg + u * 8, bl + u * 8);
    }
  };

#pragma unroll
  for (int mt = 0; mt < 2; ++mt)
#pragma unroll
    for (int nt = 0; nt < 4; ++nt) acc[mt][nt] = {};

  stage(0, 0);
  for (int k0 = 0; k0 < Dm; k0 += 64) {
    const int cur = (k0 >> 6) & 1;
    // Uniform pipeline: last iteration re-stages slab 0 into the dead buffer
    // (never consumed; s_endpgm's implicit wait-idle drains it).
    stage(cur ^ 1, (k0 + 64) & (Dm - 1));
    WAIT_ASYNC(0x8);           // oldest 8 ops (= current buffer) complete
    __syncthreads();

#pragma unroll
    for (int kc = 0; kc < 2; ++kc) {
      v16h af[2];
#pragma unroll
      for (int mt = 0; mt < 2; ++mt) {
        const _Float16* ap = &sA[cur][(wm * 32 + mt * 16 + l16) * 64 + kc * 32];
        v8h lo = *(const v8h*)&ap[k0l];
        v8h hi = *(const v8h*)&ap[k0l + 16];
#pragma unroll
        for (int i = 0; i < 8; ++i) { af[mt][i] = lo[i]; af[mt][i + 8] = hi[i]; }
      }
      v16h bf[4];
#pragma unroll
      for (int nt = 0; nt < 4; ++nt)
#pragma unroll
        for (int i = 0; i < 16; ++i)
          bf[nt][i] = sB[cur][(kc * 32 + k0l + (i & 7) + ((i >> 3) << 4)) * 128 + wn * 64 + nt * 16 + l16];

#pragma unroll
      for (int mt = 0; mt < 2; ++mt)
#pragma unroll
        for (int nt = 0; nt < 4; ++nt)
          acc[mt][nt] = WMMA(af[mt], bf[nt], acc[mt][nt]);
    }
    __syncthreads();  // all reads of 'cur' done before it is re-staged
  }
}

// ---------------------------------------------------------------------------
// Kernel 3: QKV projection. blockIdx.z selects {Q,K,V}.
// Output f16 in [B,H,S,DH] layout for attention.
// ---------------------------------------------------------------------------
__global__ void gemm_qkv_kernel(const _Float16* __restrict__ xn,
                                const _Float16* __restrict__ wq,
                                const _Float16* __restrict__ wk,
                                const _Float16* __restrict__ wv,
                                const float* __restrict__ bq,
                                const float* __restrict__ bk,
                                const float* __restrict__ bv,
                                _Float16* __restrict__ q,
                                _Float16* __restrict__ k,
                                _Float16* __restrict__ v) {
  const _Float16* W; const float* bias; _Float16* dst;
  if      (blockIdx.z == 0) { W = wq; bias = bq; dst = q; }
  else if (blockIdx.z == 1) { W = wk; bias = bk; dst = k; }
  else                      { W = wv; bias = bv; dst = v; }

  const int blockM = blockIdx.x * 128;
  const int blockN = blockIdx.y * 128;
  v8f acc[2][4];
  gemm_tile_128x128(xn, W, blockM, blockN, acc);

  const int lane = threadIdx.x & 31, wid = threadIdx.x >> 5;
  const int wm = wid & 3, wn = wid >> 2;
  const int half = lane >> 4, l16 = lane & 15;
#pragma unroll
  for (int mt = 0; mt < 2; ++mt)
#pragma unroll
    for (int nt = 0; nt < 4; ++nt)
#pragma unroll
      for (int j = 0; j < 8; ++j) {
        const int row = blockM + wm * 32 + mt * 16 + j + 8 * half;
        const int col = blockN + wn * 64 + nt * 16 + l16;
        const float val = acc[mt][nt][j] + bias[col];
        const int b = row >> 11, s = row & (Sq - 1);
        const int h = col >> 6,  d = col & (DH - 1);
        dst[((((size_t)b * Hn + h) * Sq) + s) * DH + d] = (_Float16)val;
      }
}

// ---------------------------------------------------------------------------
// Kernel 4: flash attention. Grid (S/64, B*H), 128 threads = 4 waves.
// Wave w owns 16 query rows. Online softmax, f32 accumulators.
// K staged transposed (manual), V staged via async global->LDS.
// ---------------------------------------------------------------------------
__global__ void flash_attn_kernel(const _Float16* __restrict__ Q,
                                  const _Float16* __restrict__ K,
                                  const _Float16* __restrict__ V,
                                  const unsigned char* __restrict__ mask,
                                  _Float16* __restrict__ O) {
  const int qb  = blockIdx.x;           // query block (64 rows)
  const int bh  = blockIdx.y;           // 0..63
  const int b   = bh >> 4, h = bh & 15;
  const int tid = threadIdx.x;          // 0..127
  const int lane = tid & 31, w = tid >> 5;
  const int half = lane >> 4, l16 = lane & 15, k0l = half * 8;

  __shared__ __attribute__((aligned(32))) _Float16 Kt[64 * 64];   // [dh][key]
  __shared__ __attribute__((aligned(32))) _Float16 Vl[64 * 64];   // [key][dh]
  __shared__ __attribute__((aligned(32))) _Float16 Pb[4][16 * 64];

  const size_t base = (size_t)bh * Sq * DH;
  const int srow = qb * 64 + w * 16 + l16;   // this lane's A-matrix row

  // Load Q fragments once (A-matrix 16x64 -> two 16x32 frags)
  v16h qf[2];
#pragma unroll
  for (int c = 0; c < 2; ++c) {
    const _Float16* qp = Q + base + (size_t)srow * DH + c * 32 + k0l;
    v8h lo = *(const v8h*)qp;
    v8h hi = *(const v8h*)(qp + 16);
#pragma unroll
    for (int i = 0; i < 8; ++i) { qf[c][i] = lo[i]; qf[c][i + 8] = hi[i]; }
  }

  v8f o[4]; float m[8], l[8];
#pragma unroll
  for (int t = 0; t < 4; ++t) o[t] = {};
#pragma unroll
  for (int j = 0; j < 8; ++j) { m[j] = -1e30f; l[j] = 0.f; }

  for (int kb = 0; kb < Sq / 64; ++kb) {
    // Stage V (async, natural layout) and K (transposed, via VGPRs)
    {
      const int key = tid >> 1, seg = (tid & 1) * 32;
      const _Float16* vp = V + base + (size_t)(kb * 64 + key) * DH + seg;
      _Float16*       vl = &Vl[key * 64 + seg];
#pragma unroll
      for (int u = 0; u < 4; ++u) async_copy_b128(vp + u * 8, vl + u * 8);

      const _Float16* kp = K + base + (size_t)(kb * 64 + key) * DH + seg;
#pragma unroll
      for (int j = 0; j < 32; ++j) Kt[(seg + j) * 64 + key] = kp[j];
    }
    WAIT_ASYNC(0x0);
    __syncthreads();

    // Scores S = Q * K^T  (4 key tiles of 16)
    v8f s[4];
#pragma unroll
    for (int t = 0; t < 4; ++t) {
      v8f cz = {};
      v16h b0, b1;
#pragma unroll
      for (int i = 0; i < 16; ++i) {
        const int kk = k0l + (i & 7) + ((i >> 3) << 4);
        b0[i] = Kt[(0 * 32 + kk) * 64 + t * 16 + l16];
        b1[i] = Kt[(1 * 32 + kk) * 64 + t * 16 + l16];
      }
      cz = WMMA(qf[0], b0, cz);
      cz = WMMA(qf[1], b1, cz);
      s[t] = cz;
    }

    // Scale + mask + row max (rows = VGPR idx; cols = lanes within half)
    float rmax[8];
#pragma unroll
    for (int j = 0; j < 8; ++j) rmax[j] = -1e30f;
#pragma unroll
    for (int t = 0; t < 4; ++t) {
      const int col = kb * 64 + t * 16 + l16;
      const float madd = mask[(size_t)b * Sq + col] ? -1e9f : 0.f;
#pragma unroll
      for (int j = 0; j < 8; ++j) {
        float x = s[t][j] * SCALE + madd;
        s[t][j] = x;
        rmax[j] = fmaxf(rmax[j], x);
      }
    }
#pragma unroll
    for (int j = 0; j < 8; ++j)
      for (int off = 8; off > 0; off >>= 1)
        rmax[j] = fmaxf(rmax[j], __shfl_xor(rmax[j], off, 16));

    float alpha[8], rsum[8];
#pragma unroll
    for (int j = 0; j < 8; ++j) {
      const float mn = fmaxf(m[j], rmax[j]);
      alpha[j] = __expf(m[j] - mn);
      m[j] = mn;
      rsum[j] = 0.f;
    }
#pragma unroll
    for (int t = 0; t < 4; ++t)
#pragma unroll
      for (int j = 0; j < 8; ++j) {
        const float p = __expf(s[t][j] - m[j]);
        s[t][j] = p;
        rsum[j] += p;
      }
#pragma unroll
    for (int j = 0; j < 8; ++j) {
      for (int off = 8; off > 0; off >>= 1)
        rsum[j] += __shfl_xor(rsum[j], off, 16);
      l[j] = l[j] * alpha[j] + rsum[j];
    }
#pragma unroll
    for (int t = 0; t < 4; ++t)
#pragma unroll
      for (int j = 0; j < 8; ++j) o[t][j] *= alpha[j];

    // Re-layout P: C-fragment -> per-wave LDS -> A-fragment
    _Float16* P = &Pb[w][0];
#pragma unroll
    for (int t = 0; t < 4; ++t)
#pragma unroll
      for (int j = 0; j < 8; ++j)
        P[(j + 8 * half) * 64 + t * 16 + l16] = (_Float16)s[t][j];

    v16h pf[2];
#pragma unroll
    for (int c = 0; c < 2; ++c) {
      const _Float16* pp = &P[l16 * 64 + c * 32 + k0l];
      v8h lo = *(const v8h*)pp;
      v8h hi = *(const v8h*)(pp + 16);
#pragma unroll
      for (int i = 0; i < 8; ++i) { pf[c][i] = lo[i]; pf[c][i + 8] = hi[i]; }
    }

    // O += P * V  (4 dh tiles of 16)
#pragma unroll
    for (int t = 0; t < 4; ++t) {
      v16h v0, v1;
#pragma unroll
      for (int i = 0; i < 16; ++i) {
        const int kk = k0l + (i & 7) + ((i >> 3) << 4);
        v0[i] = Vl[(0 * 32 + kk) * 64 + t * 16 + l16];
        v1[i] = Vl[(1 * 32 + kk) * 64 + t * 16 + l16];
      }
      o[t] = WMMA(pf[0], v0, o[t]);
      o[t] = WMMA(pf[1], v1, o[t]);
    }
    __syncthreads();
  }

  // Normalize and store f16 in [B*S, H*DH] row-major for the output GEMM
  float inv[8];
#pragma unroll
  for (int j = 0; j < 8; ++j) inv[j] = 1.0f / l[j];
#pragma unroll
  for (int t = 0; t < 4; ++t)
#pragma unroll
    for (int j = 0; j < 8; ++j) {
      const int srow_out = qb * 64 + w * 16 + j + 8 * half;
      const int col      = h * DH + t * 16 + l16;
      O[((size_t)b * Sq + srow_out) * Dm + col] = (_Float16)(o[t][j] * inv[j]);
    }
}

// ---------------------------------------------------------------------------
// Kernel 5: output projection + bias + residual, fp32 out
// ---------------------------------------------------------------------------
__global__ void gemm_out_kernel(const _Float16* __restrict__ attn,
                                const _Float16* __restrict__ wo,
                                const float* __restrict__ bo,
                                const float* __restrict__ resid,
                                float* __restrict__ out) {
  const int blockM = blockIdx.x * 128;
  const int blockN = blockIdx.y * 128;
  v8f acc[2][4];
  gemm_tile_128x128(attn, wo, blockM, blockN, acc);

  const int lane = threadIdx.x & 31, wid = threadIdx.x >> 5;
  const int wm = wid & 3, wn = wid >> 2;
  const int half = lane >> 4, l16 = lane & 15;
#pragma unroll
  for (int mt = 0; mt < 2; ++mt)
#pragma unroll
    for (int nt = 0; nt < 4; ++nt)
#pragma unroll
      for (int j = 0; j < 8; ++j) {
        const int row = blockM + wm * 32 + mt * 16 + j + 8 * half;
        const int col = blockN + wn * 64 + nt * 16 + l16;
        const size_t idx = (size_t)row * Dm + col;
        out[idx] = acc[mt][nt][j] + bo[col] + resid[idx];
      }
}

// ---------------------------------------------------------------------------
// Host launcher
// ---------------------------------------------------------------------------
extern "C" void kernel_launch(void* const* d_in, const int* in_sizes, int n_in,
                              void* d_out, int out_size, void* d_ws, size_t ws_size,
                              hipStream_t stream) {
  const float*         hs   = (const float*)d_in[0];
  const unsigned char* mask = (const unsigned char*)d_in[1];
  const float* Wq = (const float*)d_in[2];
  const float* bq = (const float*)d_in[3];
  const float* Wk = (const float*)d_in[4];
  const float* bk = (const float*)d_in[5];
  const float* Wv = (const float*)d_in[6];
  const float* bv = (const float*)d_in[7];
  const float* Wo = (const float*)d_in[8];
  const float* bo = (const float*)d_in[9];
  const float* g  = (const float*)d_in[10];
  const float* be = (const float*)d_in[11];
  float* out = (float*)d_out;

  // Workspace carve (all f16):
  //   xn[8192*1024] | wq,wk,wv,wo[1024*1024]*4 | q,k,v[B*H*S*DH]*3 | attn[8192*1024]
  char* ws = (char*)d_ws;
  _Float16* xn   = (_Float16*)ws; ws += (size_t)Mtot * Dm * 2;
  _Float16* wq16 = (_Float16*)ws; ws += (size_t)Dm * Dm * 2;
  _Float16* wk16 = (_Float16*)ws; ws += (size_t)Dm * Dm * 2;
  _Float16* wv16 = (_Float16*)ws; ws += (size_t)Dm * Dm * 2;
  _Float16* wo16 = (_Float16*)ws; ws += (size_t)Dm * Dm * 2;
  _Float16* q    = (_Float16*)ws; ws += (size_t)Bsz * Hn * Sq * DH * 2;
  _Float16* k    = (_Float16*)ws; ws += (size_t)Bsz * Hn * Sq * DH * 2;
  _Float16* v    = (_Float16*)ws; ws += (size_t)Bsz * Hn * Sq * DH * 2;
  _Float16* attn = (_Float16*)ws; ws += (size_t)Mtot * Dm * 2;

  ln_cast_kernel<<<Mtot, 256, 0, stream>>>(hs, g, be, xn);

  const int wn = Dm * Dm;
  cast_f32_f16<<<(wn + 255) / 256, 256, 0, stream>>>(Wq, wq16, wn);
  cast_f32_f16<<<(wn + 255) / 256, 256, 0, stream>>>(Wk, wk16, wn);
  cast_f32_f16<<<(wn + 255) / 256, 256, 0, stream>>>(Wv, wv16, wn);
  cast_f32_f16<<<(wn + 255) / 256, 256, 0, stream>>>(Wo, wo16, wn);

  gemm_qkv_kernel<<<dim3(Mtot / 128, Dm / 128, 3), 256, 0, stream>>>(
      xn, wq16, wk16, wv16, bq, bk, bv, q, k, v);

  flash_attn_kernel<<<dim3(Sq / 64, Bsz * Hn), 128, 0, stream>>>(
      q, k, v, mask, attn);

  gemm_out_kernel<<<dim3(Mtot / 128, Dm / 128), 256, 0, stream>>>(
      attn, wo16, bo, hs, out);
}
